// HeterdenseGAT_74732430950522
// MI455X (gfx1250) — matrix operand order repaired and checked by reading, not compiled, not run on hardware
//
#include <hip/hip_runtime.h>
#include <hip/hip_bf16.h>
#include <stdint.h>

// ---- problem constants (match reference) ----
#define NN      6144
#define NUSER   4096
#define RR      2
#define KK      3
#define FIN     25
#define FOUT    64
#define DD2     64
#define NCLS    2

typedef __attribute__((ext_vector_type(16))) _Float16 v16h;
typedef __attribute__((ext_vector_type(8)))  float    v8f;

// ------------------------------------------------------------------
// Kernel 0: hp[rk] = h @ W[rk]  (stored transposed, f16, for WMMA B),
//           s[rk][n] = hp[n,:].a_src,  d[rk][n] = hp[n,:].a_dst
// grid: (N/4, 6), block: 256  (64 threads per node, 4 nodes per block)
// ------------------------------------------------------------------
__global__ void gat_hp_sd_kernel(const float* __restrict__ h,
                                 const float* __restrict__ w,
                                 const float* __restrict__ asrc,
                                 const float* __restrict__ adst,
                                 _Float16* __restrict__ hpT,
                                 float* __restrict__ s,
                                 float* __restrict__ d) {
    const int rk  = blockIdx.y;
    const int tid = threadIdx.x;
    const int ni  = tid >> 6;          // node within block (0..3)
    const int o   = tid & 63;          // output feature
    const int n   = blockIdx.x * 4 + ni;

    __shared__ float hs[4][FIN];
    __shared__ float hp_s[4][FOUT];

    if (tid < 4 * FIN) hs[tid / FIN][tid % FIN] =
        h[(size_t)(blockIdx.x * 4 + tid / FIN) * FIN + (tid % FIN)];
    __syncthreads();

    float acc = 0.f;
    #pragma unroll
    for (int f = 0; f < FIN; ++f)
        acc += hs[ni][f] * w[((size_t)rk * FIN + f) * FOUT + o];

    hpT[((size_t)rk * FOUT + o) * NN + n] = (_Float16)acc;
    hp_s[ni][o] = acc;
    __syncthreads();

    if (tid < 8) {                      // 4 nodes x {src,dst}
        const int nj    = tid >> 1;
        const int which = tid & 1;
        const float* av = which ? (adst + (size_t)rk * FOUT)
                                : (asrc + (size_t)rk * FOUT);
        float sum = 0.f;
        #pragma unroll 8
        for (int oo = 0; oo < FOUT; ++oo) sum += hp_s[nj][oo] * av[oo];
        float* dst = which ? d : s;
        dst[(size_t)rk * NN + blockIdx.x * 4 + nj] = sum;
    }
}

// ------------------------------------------------------------------
// Kernel 1: masked row max.  mx[rk][n] = lrelu(s[rk][n] + max_{adj} d[rk][m])
// one wave per (r,n) row; block = 256 (8 waves); grid = R*N/8
// ------------------------------------------------------------------
__global__ void gat_rowmax_kernel(const uint8_t* __restrict__ adj,
                                  const float* __restrict__ s,
                                  const float* __restrict__ d,
                                  float* __restrict__ mx) {
    const int gid  = blockIdx.x * 8 + (threadIdx.x >> 5);
    const int lane = threadIdx.x & 31;
    const int r = gid / NN;
    const int n = gid % NN;

    const uint64_t* arow =
        (const uint64_t*)(adj + (size_t)r * NN * NN + (size_t)n * NN);

    float dm0 = -3.0e38f, dm1 = -3.0e38f, dm2 = -3.0e38f;
    const float* d0p = d + (size_t)(r * 3 + 0) * NN;
    const float* d1p = d + (size_t)(r * 3 + 1) * NN;
    const float* d2p = d + (size_t)(r * 3 + 2) * NN;

    for (int i = 0; i < NN / 256; ++i) {              // 24 iterations
        const int m0 = i * 256 + lane * 8;
        const uint64_t bits = arow[i * 32 + lane];
        if (bits) {
            const float4* a0 = (const float4*)(d0p + m0);
            const float4* a1 = (const float4*)(d1p + m0);
            const float4* a2 = (const float4*)(d2p + m0);
            float v0[8], v1[8], v2[8];
            { float4 x = a0[0], y = a0[1];
              v0[0]=x.x; v0[1]=x.y; v0[2]=x.z; v0[3]=x.w;
              v0[4]=y.x; v0[5]=y.y; v0[6]=y.z; v0[7]=y.w; }
            { float4 x = a1[0], y = a1[1];
              v1[0]=x.x; v1[1]=x.y; v1[2]=x.z; v1[3]=x.w;
              v1[4]=y.x; v1[5]=y.y; v1[6]=y.z; v1[7]=y.w; }
            { float4 x = a2[0], y = a2[1];
              v2[0]=x.x; v2[1]=x.y; v2[2]=x.z; v2[3]=x.w;
              v2[4]=y.x; v2[5]=y.y; v2[6]=y.z; v2[7]=y.w; }
            #pragma unroll
            for (int j = 0; j < 8; ++j) {
                if ((bits >> (8 * j)) & 0xffull) {
                    dm0 = fmaxf(dm0, v0[j]);
                    dm1 = fmaxf(dm1, v1[j]);
                    dm2 = fmaxf(dm2, v2[j]);
                }
            }
        }
    }
    #pragma unroll
    for (int off = 16; off > 0; off >>= 1) {
        dm0 = fmaxf(dm0, __shfl_xor(dm0, off, 32));
        dm1 = fmaxf(dm1, __shfl_xor(dm1, off, 32));
        dm2 = fmaxf(dm2, __shfl_xor(dm2, off, 32));
    }
    if (lane == 0) {
        float dm[3] = {dm0, dm1, dm2};
        #pragma unroll
        for (int k = 0; k < 3; ++k) {
            const size_t idx = (size_t)(r * 3 + k) * NN + n;
            const float e = s[idx] + dm[k];
            mx[idx] = fmaxf(e, 0.2f * e);            // leaky_relu, slope 0.2
        }
    }
}

// ------------------------------------------------------------------
// Kernel 2: flash-style masked softmax + P*V with WMMA.
// block = 96 (3 waves, one per head k); each block: 16 rows of relation r.
// grid = (N/16, R). Output emb[r][n][o] = mean_k(attn_k @ hp_k) + bias[r][o].
// ------------------------------------------------------------------
__global__ void gat_attn_wmma_kernel(const uint8_t* __restrict__ adj,
                                     const _Float16* __restrict__ hpT,
                                     const float* __restrict__ s,
                                     const float* __restrict__ d,
                                     const float* __restrict__ mx,
                                     const float* __restrict__ bias,
                                     float* __restrict__ emb) {
    const int r    = blockIdx.y;
    const int n0   = blockIdx.x * 16;
    const int wave = threadIdx.x >> 5;    // head k
    const int lane = threadIdx.x & 31;
    const int rk   = r * 3 + wave;

    const int mrow = lane & 15;           // row of A this lane supplies
    const int half = lane >> 4;           // 0 or 1 (ISA f16 A-layout lane half)
    const int rown = n0 + mrow;

    const float s_row  = s [(size_t)rk * NN + rown];
    const float mx_row = mx[(size_t)rk * NN + rown];

    const uint8_t* arow = adj + (size_t)r * NN * NN + (size_t)rown * NN;
    const float*   drow = d + (size_t)rk * NN;
    const size_t   hbase = (size_t)(rk * FOUT + mrow) * NN;   // column o = t*16+mrow

    v8f acc[4] = {v8f{}, v8f{}, v8f{}, v8f{}};
    float psum = 0.f;
    const int kb = half * 16;             // K sub-base: lanes 0-15 -> 0, 16-31 -> 16

    for (int c0 = 0; c0 < NN; c0 += 32) {
        // prefetch next adjacency tile (CDNA5 global_prefetch path)
        __builtin_prefetch(arow + c0 + 128, 0, 0);

        // adjacency bytes for this lane's 16 K-slots
        const uint64_t b_lo = *(const uint64_t*)(arow + c0 + kb);
        const uint64_t b_hi = *(const uint64_t*)(arow + c0 + kb + 8);

        // d values for the same 16 K-slots (two aligned float4 pairs)
        const float4* dp = (const float4*)(drow + c0 + kb);
        float dv[16];
        { float4 x = dp[0], y = dp[1];
          dv[0]=x.x; dv[1]=x.y; dv[2]=x.z; dv[3]=x.w;
          dv[4]=y.x; dv[5]=y.y; dv[6]=y.z; dv[7]=y.w; }
        { float4 x = dp[2], y = dp[3];
          dv[8]=x.x;  dv[9]=x.y;  dv[10]=x.z; dv[11]=x.w;
          dv[12]=y.x; dv[13]=y.y; dv[14]=y.z; dv[15]=y.w; }

        // build A fragment: p = adj ? exp(lrelu(s+d) - mx) : 0
        v16h a;
        #pragma unroll
        for (int j = 0; j < 16; ++j) {
            const unsigned byte = (unsigned)(((j < 8) ? (b_lo >> (8 * j))
                                                      : (b_hi >> (8 * (j - 8)))) & 0xffull);
            float e = s_row + dv[j];
            e = fmaxf(e, 0.2f * e);
            const float p = byte ? __expf(e - mx_row) : 0.0f;
            psum += p;
            a[j] = (_Float16)p;
        }

        // B fragments: 16 consecutive K-rows of one hpT column, contiguous f16
        const size_t mb = (size_t)(c0 + half * 16);
        const v16h b0 = *(const v16h*)(hpT + hbase + 0 * 16 * NN + mb);
        acc[0] = __builtin_amdgcn_wmma_f32_16x16x32_f16(false, a, false, b0,
                                                        (short)0, acc[0], false, false);
        const v16h b1 = *(const v16h*)(hpT + hbase + 1 * 16 * NN + mb);
        acc[1] = __builtin_amdgcn_wmma_f32_16x16x32_f16(false, a, false, b1,
                                                        (short)0, acc[1], false, false);
        const v16h b2 = *(const v16h*)(hpT + hbase + 2 * 16 * NN + mb);
        acc[2] = __builtin_amdgcn_wmma_f32_16x16x32_f16(false, a, false, b2,
                                                        (short)0, acc[2], false, false);
        const v16h b3 = *(const v16h*)(hpT + hbase + 3 * 16 * NN + mb);
        acc[3] = __builtin_amdgcn_wmma_f32_16x16x32_f16(false, a, false, b3,
                                                        (short)0, acc[3], false, false);
    }

    // row sums: lanes l and l^16 hold the two halves of row (l&15)
    psum += __shfl_xor(psum, 16, 32);
    const float inv = 1.0f / psum;

    __shared__ float rn_s[3][16];
    __shared__ float outs[3][16 * FOUT];
    if (lane < 16) rn_s[wave][lane] = inv;
    __syncthreads();

    // scatter scaled C fragments (C layout: row = v + 8*(lane>=16), col = lane%16)
    const int rb = half * 8;
    #pragma unroll
    for (int t = 0; t < 4; ++t) {
        #pragma unroll
        for (int v = 0; v < 8; ++v) {
            const int row = v + rb;
            const int col = t * 16 + mrow;
            outs[wave][row * FOUT + col] = acc[t][v] * rn_s[wave][row];
        }
    }
    __syncthreads();

    for (int i = threadIdx.x; i < 16 * FOUT; i += 96) {
        const float val = (outs[0][i] + outs[1][i] + outs[2][i]) * (1.0f / 3.0f)
                        + bias[r * FOUT + (i & 63)];
        emb[((size_t)r * NN + n0 + (i >> 6)) * FOUT + (i & 63)] = val;
    }
}

// ------------------------------------------------------------------
// Kernel 3: semantic attention fusion + FC + log_softmax.
// block = 64 threads per node, grid = N_USER.
// ------------------------------------------------------------------
__global__ void gat_fuse_kernel(const float* __restrict__ h,
                                const float* __restrict__ emb,
                                const float* __restrict__ aw1,
                                const float* __restrict__ aw2,
                                const float* __restrict__ am,
                                const float* __restrict__ fcw,
                                const float* __restrict__ fcb,
                                float* __restrict__ out) {
    const int n   = blockIdx.x;
    const int tid = threadIdx.x;

    __shared__ float hsh[FIN];
    __shared__ float ta[RR][FOUT];
    __shared__ float red[RR][FOUT];
    __shared__ float red2[NCLS][FOUT];
    __shared__ float bls[RR];
    __shared__ float lg[NCLS];

    if (tid < FIN) hsh[tid] = h[(size_t)n * FIN + tid];
    ta[0][tid] = emb[((size_t)0 * NN + n) * FOUT + tid];
    ta[1][tid] = emb[((size_t)1 * NN + n) * FOUT + tid];
    __syncthreads();

    float q1 = 0.f;
    #pragma unroll
    for (int f = 0; f < FIN; ++f) q1 += hsh[f] * aw1[f * DD2 + tid];

    #pragma unroll
    for (int rr = 0; rr < RR; ++rr) {
        float a = q1;
        #pragma unroll 8
        for (int o = 0; o < FOUT; ++o) a += ta[rr][o] * aw2[o * DD2 + tid];
        red[rr][tid] = tanhf(a) * am[tid];
    }
    __syncthreads();

    if (tid < RR) {
        float sum = 0.f;
        #pragma unroll 8
        for (int i = 0; i < DD2; ++i) sum += red[tid][i];
        bls[tid] = sum;
    }
    __syncthreads();

    const float b0l = bls[0], b1l = bls[1];
    const float bm  = fmaxf(b0l, b1l);
    const float e0  = __expf(b0l - bm), e1 = __expf(b1l - bm);
    const float biv = 1.0f / (e0 + e1);
    const float beta0 = e0 * biv, beta1 = e1 * biv;
    const float fus = beta0 * ta[0][tid] + beta1 * ta[1][tid];

    #pragma unroll
    for (int c = 0; c < NCLS; ++c)
        red2[c][tid] = ta[0][tid] * fcw[tid * NCLS + c]
                     + ta[1][tid] * fcw[(FOUT + tid) * NCLS + c]
                     + fus        * fcw[(2 * FOUT + tid) * NCLS + c];
    __syncthreads();

    if (tid < NCLS) {
        float sum = 0.f;
        #pragma unroll 8
        for (int i = 0; i < FOUT; ++i) sum += red2[tid][i];
        lg[tid] = sum + fcb[tid];
    }
    __syncthreads();

    if (tid == 0) {
        const float l0 = lg[0], l1 = lg[1];
        const float m  = fmaxf(l0, l1);
        const float ls = m + logf(__expf(l0 - m) + __expf(l1 - m));
        out[(size_t)n * NCLS + 0] = l0 - ls;
        out[(size_t)n * NCLS + 1] = l1 - ls;
    }
}

// ------------------------------------------------------------------
// Launch. Workspace layout (bytes, all offsets 32B-aligned):
//   hpT : 6*64*6144 f16  = 4,718,592
//   s   : 6*6144 f32     =   147,456
//   d   : 6*6144 f32     =   147,456
//   mx  : 6*6144 f32     =   147,456
//   emb : 2*6144*64 f32  = 3,145,728     total ~8.3 MB
// ------------------------------------------------------------------
extern "C" void kernel_launch(void* const* d_in, const int* in_sizes, int n_in,
                              void* d_out, int out_size, void* d_ws, size_t ws_size,
                              hipStream_t stream) {
    const uint8_t* adj  = (const uint8_t*)d_in[0];
    const float*   h    = (const float*)d_in[1];
    const float*   w    = (const float*)d_in[2];
    const float*   asrc = (const float*)d_in[3];
    const float*   adst = (const float*)d_in[4];
    const float*   bias = (const float*)d_in[5];
    const float*   aw1  = (const float*)d_in[6];
    const float*   aw2  = (const float*)d_in[7];
    const float*   am   = (const float*)d_in[8];
    const float*   fcw  = (const float*)d_in[9];
    const float*   fcb  = (const float*)d_in[10];
    float* out = (float*)d_out;

    char* ws = (char*)d_ws;
    _Float16* hpT = (_Float16*)(ws);
    float* s   = (float*)(ws + 4718592);
    float* d   = (float*)(ws + 4718592 + 147456);
    float* mx  = (float*)(ws + 4718592 + 2 * 147456);
    float* emb = (float*)(ws + 4718592 + 3 * 147456);

    gat_hp_sd_kernel<<<dim3(NN / 4, RR * KK), 256, 0, stream>>>(
        h, w, asrc, adst, hpT, s, d);

    gat_rowmax_kernel<<<dim3(RR * NN / 8), 256, 0, stream>>>(adj, s, d, mx);

    gat_attn_wmma_kernel<<<dim3(NN / 16, RR), 96, 0, stream>>>(
        adj, hpT, s, d, mx, bias, emb);

    gat_fuse_kernel<<<dim3(NUSER), 64, 0, stream>>>(
        h, emb, aw1, aw2, am, fcw, fcb, out);
}
